// GroupedFeedForward_13709535609205
// MI455X (gfx1250) — matrix operand order repaired and checked by reading, not compiled, not run on hardware
//
#include <hip/hip_runtime.h>

// Problem dims (fixed by the reference)
#define G_ 8
#define T_ 2048
#define D_ 2048
#define H_ 5632

typedef __attribute__((ext_vector_type(16))) __bf16 v16bf;
typedef __attribute__((ext_vector_type(8)))  __bf16 v8bf;
typedef __attribute__((ext_vector_type(4)))  __bf16 v4bf;
typedef __attribute__((ext_vector_type(8)))  float  v8f;
typedef __attribute__((ext_vector_type(4)))  float  v4f;
typedef __attribute__((ext_vector_type(4)))  unsigned int u32x4;
typedef __attribute__((ext_vector_type(8)))  int    i32x8;
typedef __attribute__((ext_vector_type(4)))  int    i32x4;

// LDS row stride in bf16 elements: 32 data + 8 pad = 80 bytes (keeps b128 16B-aligned,
// staggers banks between consecutive rows; matches TDM pad_interval=16dw/pad_amount=4dw).
#define LDS_STRIDE 40
#define KSTEP 32
#define MTILE 128
#define NTILE 128

#if __has_builtin(__builtin_amdgcn_tensor_load_to_lds)
#define USE_TDM 1
#else
#define USE_TDM 0
#endif

static __device__ __forceinline__ v8f bf16_wmma(v16bf a, v16bf b, v8f c) {
  // D = A(16x32 bf16) x B(32x16 bf16) + C(16x16 f32)
  return __builtin_amdgcn_wmma_f32_16x16x32_bf16(
      /*neg_a=*/false, a, /*neg_b=*/false, b,
      /*c_mod=*/(short)0, c, /*reuse_a=*/false, /*reuse_b=*/false);
}

// ---- LDS fragment loads (match ISA 7.12.2 16-bit layouts) -------------------

// A fragment: 16x32 (MxK). lane<16: row=lane, K runs {0..7,16..23};
// lane>=16: row=lane-16, K runs {8..15,24..31}. Two 16B ds loads.
static __device__ __forceinline__ v16bf load_a_frag(const __bf16* s, int m0, int lane) {
  int row  = m0 + (lane & 15);
  int koff = (lane >> 4) * 8;
  const __bf16* p = s + row * LDS_STRIDE + koff;
  v8bf lo = *(const v8bf*)(p);        // K = koff .. koff+7
  v8bf hi = *(const v8bf*)(p + 16);   // K = koff+16 .. koff+23
  return __builtin_shufflevector(lo, hi, 0,1,2,3,4,5,6,7,8,9,10,11,12,13,14,15);
}

// B fragment: 32x16 (KxN) from LDS stored transposed [n][k].
// lane<16: col=lane, K runs {0..7,16..23}; lane>=16: col=lane-16, K runs {8..15,24..31}. Two 16B ds loads.
static __device__ __forceinline__ v16bf load_b_frag(const __bf16* s, int n0, int lane) {
  int col  = n0 + (lane & 15);
  int koff = (lane >> 4) * 8;
  const __bf16* p = s + col * LDS_STRIDE + koff;
  v8bf lo = *(const v8bf*)(p);        // K = koff .. koff+7
  v8bf hi = *(const v8bf*)(p + 16);   // K = koff+16 .. koff+23
  return __builtin_shufflevector(lo, hi, 0,1,2,3,4,5,6,7,8,9,10,11,12,13,14,15);
}

// ---- Global -> LDS tile loaders (f32 -> bf16 conversion at store time) ------

// A tile: [128 m][32 k] from row-major f32 src (row stride ld). Coalesced
// float4 reads along k; packed v4bf (b64) LDS stores.
static __device__ __forceinline__ void load_a_tile_f32(__bf16* s, const float* src,
                                                       int m_base, int k_base,
                                                       int tid, int ld) {
  int c  = tid & 7;        // float4 chunk in row: k0 = c*4
  int r0 = tid >> 3;       // 0..31
#pragma unroll
  for (int it = 0; it < 4; ++it) {
    int r = r0 + it * 32;
    v4f v = *(const v4f*)(src + (size_t)(m_base + r) * ld + k_base + c * 4);
    v4bf b = { (__bf16)v[0], (__bf16)v[1], (__bf16)v[2], (__bf16)v[3] };
    *(v4bf*)(s + r * LDS_STRIDE + c * 4) = b;   // 8B aligned
  }
}

// B tile: source is [K][N] row-major f32 (row stride ld). Stored transposed to
// LDS as [n][k]. Lanes span n (coalesced 4B reads), each thread gathers 16 k
// values for one column in 4 packed b64 LDS stores.
static __device__ __forceinline__ void load_b_tile_f32_tr(__bf16* s, const float* src,
                                                          int k_base, int n_base,
                                                          int tid, int ld) {
  int n     = tid & 127;
  int khalf = tid >> 7;    // 0..1
  const float* col = src + (size_t)k_base * ld + n_base + n;
#pragma unroll
  for (int c = 0; c < 4; ++c) {
    int k0 = khalf * 16 + c * 4;
    float f0 = col[(size_t)(k0 + 0) * ld];
    float f1 = col[(size_t)(k0 + 1) * ld];
    float f2 = col[(size_t)(k0 + 2) * ld];
    float f3 = col[(size_t)(k0 + 3) * ld];
    v4bf b = { (__bf16)f0, (__bf16)f1, (__bf16)f2, (__bf16)f3 };
    *(v4bf*)(s + n * LDS_STRIDE + k0) = b;      // 8B aligned
  }
}

#if !USE_TDM
// Fallback: A tile already bf16 in memory (workspace h): [128 m][32 k], b128 copies.
static __device__ __forceinline__ void load_a_tile_bf16(__bf16* s, const __bf16* src,
                                                        int m_base, int k_base,
                                                        int tid, int ld) {
  int c  = tid & 3;        // 8-elem chunk: k0 = c*8
  int r0 = tid >> 2;       // 0..63
#pragma unroll
  for (int it = 0; it < 2; ++it) {
    int r = r0 + it * 64;
    v8bf v = *(const v8bf*)(src + (size_t)(m_base + r) * ld + k_base + c * 8);
    *(v8bf*)(s + r * LDS_STRIDE + c * 8) = v;   // 16B aligned
  }
}
#else
// TDM copy of the bf16 A tile: one tensor_load_to_lds moves a 128x32 bf16 tile
// (rows of 64B) into LDS, inserting 16B padding after every 64B row so the LDS
// image has exactly LDS_STRIDE(40 bf16)=80B rows — identical layout to the
// manual loader. Issued by one wave; completion via s_wait_tensorcnt.
// D# packing per CDNA5 ISA.
// This toolchain exposes the 6-arg builtin:
//   (u32x4 g0, i32x8 g1, i32x4 g2, i32x4 g3, i32x8 extra, i32 cpol)
static __device__ __forceinline__ void tdm_load_a_tile(unsigned int lds_byte_addr,
                                                       const __bf16* src,
                                                       int m_base, int k_base, int ld) {
  unsigned long long ga =
      (unsigned long long)(uintptr_t)(src + (size_t)m_base * ld + k_base);
  u32x4 g0;
  g0[0] = 1u;                                   // count=1 valid user descriptor
  g0[1] = lds_byte_addr;                        // lds_addr (bytes)
  g0[2] = (unsigned int)ga;                     // global_addr[31:0]
  g0[3] = (unsigned int)((ga >> 32) & 0x01FFFFFFu) | (2u << 30); // ga[56:32] | type=2

  i32x8 g1;
  // data_size=1 (2B elems) | pad_enable | pad_interval=3 (16 dw = 64B)
  // | pad_amount=3 (4 dw = 16B)
  g1[0] = (int)((1u << 16) | (1u << 20) | (3u << 22) | (3u << 25));
  g1[1] = (int)(((unsigned int)ld & 0xFFFFu) << 16);   // tensor_dim0[15:0] = ld
  g1[2] = (int)((((unsigned int)ld >> 16) & 0xFFFFu) | (((unsigned int)T_) << 16)); // dim0 hi | tensor_dim1 lo
  g1[3] = (int)(32u << 16);                            // tensor_dim1 hi=0 | tile_dim0=32 elems
  g1[4] = 128;                                         // tile_dim1=128 rows, tile_dim2=0
  g1[5] = ld;                                          // tensor_dim0_stride[31:0]
  g1[6] = 0;                                           // stride hi / dim1_stride lo
  g1[7] = 0;

  i32x4 gz = {0, 0, 0, 0};                             // 2D tensor: groups 2/3 unused
  i32x8 gx = {0, 0, 0, 0, 0, 0, 0, 0};                 // extra group (unused): zeros
  __builtin_amdgcn_tensor_load_to_lds(g0, g1, gz, gz, gx, 0);
}
#endif

// ---- Kernel 1: h = silu(x@w1) * (x@w3), h stored bf16 ----------------------

__global__ __launch_bounds__(256)
void ffn_gate_kernel(const float* __restrict__ x, const float* __restrict__ w1,
                     const float* __restrict__ w3, __bf16* __restrict__ h) {
  __shared__ __bf16 sA [2][MTILE * LDS_STRIDE];
  __shared__ __bf16 sB1[2][NTILE * LDS_STRIDE];
  __shared__ __bf16 sB3[2][NTILE * LDS_STRIDE];

  const int g      = blockIdx.z;
  const int m_base = blockIdx.y * MTILE;
  const int n_base = blockIdx.x * NTILE;
  const float* xg  = x  + (size_t)g * T_ * D_;
  const float* w1g = w1 + (size_t)g * D_ * H_;
  const float* w3g = w3 + (size_t)g * D_ * H_;
  __bf16*      hg  = h  + (size_t)g * T_ * H_;

  const int tid  = threadIdx.x;
  const int lane = tid & 31;
  const int wave = tid >> 5;
  const int wm   = (wave >> 2) * 64;   // 0 / 64
  const int wn   = (wave & 3)  * 32;   // 0 / 32 / 64 / 96

  const v8f vzero = {0.f, 0.f, 0.f, 0.f, 0.f, 0.f, 0.f, 0.f};
  v8f acc1[4][2], acc3[4][2];
#pragma unroll
  for (int i = 0; i < 4; ++i)
#pragma unroll
    for (int j = 0; j < 2; ++j) { acc1[i][j] = vzero; acc3[i][j] = vzero; }

  // prologue
  load_a_tile_f32   (sA [0], xg , m_base, 0, tid, D_);
  load_b_tile_f32_tr(sB1[0], w1g, 0, n_base, tid, H_);
  load_b_tile_f32_tr(sB3[0], w3g, 0, n_base, tid, H_);
  __syncthreads();

  const int nk = D_ / KSTEP;  // 64
  for (int kt = 0; kt < nk; ++kt) {
    const int buf  = kt & 1;
    const int nbuf = buf ^ 1;
    if (kt + 1 < nk) {
      const int kb = (kt + 1) * KSTEP;
      load_a_tile_f32   (sA [nbuf], xg , m_base, kb, tid, D_);
      load_b_tile_f32_tr(sB1[nbuf], w1g, kb, n_base, tid, H_);
      load_b_tile_f32_tr(sB3[nbuf], w3g, kb, n_base, tid, H_);
    }

    v16bf af[4];
#pragma unroll
    for (int i = 0; i < 4; ++i) af[i] = load_a_frag(sA[buf], wm + i * 16, lane);
#pragma unroll
    for (int j = 0; j < 2; ++j) {
      v16bf b1 = load_b_frag(sB1[buf], wn + j * 16, lane);
      v16bf b3 = load_b_frag(sB3[buf], wn + j * 16, lane);
#pragma unroll
      for (int i = 0; i < 4; ++i) {
        acc1[i][j] = bf16_wmma(af[i], b1, acc1[i][j]);
        acc3[i][j] = bf16_wmma(af[i], b3, acc3[i][j]);
      }
    }
    __syncthreads();
  }

  // fused SwiGLU epilogue; C layout: elem e -> M = 8*(lane>>4)+e, N = lane&15
#pragma unroll
  for (int i = 0; i < 4; ++i)
#pragma unroll
    for (int j = 0; j < 2; ++j) {
      const int M0 = m_base + wm + i * 16 + ((lane >> 4) * 8);
      const int N0 = n_base + wn + j * 16 + (lane & 15);
#pragma unroll
      for (int e = 0; e < 8; ++e) {
        float a  = acc1[i][j][e];
        float sg = 1.0f / (1.0f + __expf(-a));
        float hv = a * sg * acc3[i][j][e];
        hg[(size_t)(M0 + e) * H_ + N0] = (__bf16)hv;
      }
    }
}

// ---- Kernel 2: out = h @ w2 (f32 out), A tile via TDM ----------------------

__global__ __launch_bounds__(256)
void ffn_out_kernel(const __bf16* __restrict__ h, const float* __restrict__ w2,
                    float* __restrict__ out) {
  __shared__ __bf16 sA[2][MTILE * LDS_STRIDE];
  __shared__ __bf16 sB[2][NTILE * LDS_STRIDE];

  const int g      = blockIdx.z;
  const int m_base = blockIdx.y * MTILE;
  const int n_base = blockIdx.x * NTILE;
  const __bf16* hg = h   + (size_t)g * T_ * H_;
  const float* w2g = w2  + (size_t)g * H_ * D_;
  float*       og  = out + (size_t)g * T_ * D_;

  const int tid  = threadIdx.x;
  const int lane = tid & 31;
  const int wave = tid >> 5;
  const int wm   = (wave >> 2) * 64;
  const int wn   = (wave & 3)  * 32;

#if USE_TDM
  const unsigned int ldsA0 = (unsigned int)(uintptr_t)(&sA[0][0]);
  const unsigned int ldsA1 = (unsigned int)(uintptr_t)(&sA[1][0]);
#endif

  const v8f vzero = {0.f, 0.f, 0.f, 0.f, 0.f, 0.f, 0.f, 0.f};
  v8f acc[4][2];
#pragma unroll
  for (int i = 0; i < 4; ++i)
#pragma unroll
    for (int j = 0; j < 2; ++j) acc[i][j] = vzero;

  // prologue
#if USE_TDM
  if (wave == 0) tdm_load_a_tile(ldsA0, hg, m_base, 0, H_);
#else
  load_a_tile_bf16(sA[0], hg, m_base, 0, tid, H_);
#endif
  load_b_tile_f32_tr(sB[0], w2g, 0, n_base, tid, D_);
#if USE_TDM
  if (wave == 0) __builtin_amdgcn_s_wait_tensorcnt(0);
#endif
  __syncthreads();

  const int nk = H_ / KSTEP;  // 176
  for (int kt = 0; kt < nk; ++kt) {
    const int buf  = kt & 1;
    const int nbuf = buf ^ 1;
    if (kt + 1 < nk) {
      const int kb = (kt + 1) * KSTEP;
#if USE_TDM
      if (wave == 0) tdm_load_a_tile(nbuf ? ldsA1 : ldsA0, hg, m_base, kb, H_);
#else
      load_a_tile_bf16(sA[nbuf], hg, m_base, kb, tid, H_);
#endif
      load_b_tile_f32_tr(sB[nbuf], w2g, kb, n_base, tid, D_);
    }

    v16bf af[4];
#pragma unroll
    for (int i = 0; i < 4; ++i) af[i] = load_a_frag(sA[buf], wm + i * 16, lane);
#pragma unroll
    for (int j = 0; j < 2; ++j) {
      v16bf bf = load_b_frag(sB[buf], wn + j * 16, lane);
#pragma unroll
      for (int i = 0; i < 4; ++i) acc[i][j] = bf16_wmma(af[i], bf, acc[i][j]);
    }
#if USE_TDM
    if (wave == 0) __builtin_amdgcn_s_wait_tensorcnt(0);
#endif
    __syncthreads();
  }

#pragma unroll
  for (int i = 0; i < 4; ++i)
#pragma unroll
    for (int j = 0; j < 2; ++j) {
      const int M0 = m_base + wm + i * 16 + ((lane >> 4) * 8);
      const int N0 = n_base + wn + j * 16 + (lane & 15);
#pragma unroll
      for (int e = 0; e < 8; ++e)
        og[(size_t)(M0 + e) * D_ + N0] = acc[i][j][e];
    }
}

// ---- Host launch ------------------------------------------------------------

extern "C" void kernel_launch(void* const* d_in, const int* in_sizes, int n_in,
                              void* d_out, int out_size, void* d_ws, size_t ws_size,
                              hipStream_t stream) {
  (void)in_sizes; (void)n_in; (void)out_size; (void)ws_size;
  const float* x  = (const float*)d_in[0];
  const float* w1 = (const float*)d_in[1];
  const float* w2 = (const float*)d_in[2];
  const float* w3 = (const float*)d_in[3];
  float* out      = (float*)d_out;
  __bf16* hbuf    = (__bf16*)d_ws;   // G*T*H bf16 intermediate (~185 MB)

  dim3 grid1(H_ / NTILE, T_ / MTILE, G_);  // 44 x 16 x 8
  ffn_gate_kernel<<<grid1, 256, 0, stream>>>(x, w1, w3, hbuf);

  dim3 grid2(D_ / NTILE, T_ / MTILE, G_);  // 16 x 16 x 8
  ffn_out_kernel<<<grid2, 256, 0, stream>>>(hbuf, w2, out);
}
